// DrugGVPModel_70471823392968
// MI455X (gfx1250) — matrix operand order repaired and checked by reading, not compiled, not run on hardware
//
#include <hip/hip_runtime.h>
#include <math.h>

// ---------------------------------------------------------------------------
// GVP-GNN forward for MI455X (gfx1250).  All matmuls routed through
// v_wmma_f32_16x16x32_bf16 (bf16 A/B converted on LDS-stage, f32 accumulate).
// Vector features kept in [n, 3, C] layout so every GVP op is a plain GEMM.
// ---------------------------------------------------------------------------

#define NNODES 100000
#define NEDGES 400000
#define NGRAPH 2000

typedef __bf16 bf16_t;
typedef bf16_t v16bf __attribute__((ext_vector_type(16)));
typedef bf16_t v8bf  __attribute__((ext_vector_type(8)));
typedef float  v8f   __attribute__((ext_vector_type(8)));

static __device__ __forceinline__ bf16_t f2bf(float f) {
  unsigned int u = __builtin_bit_cast(unsigned int, f);
  u += 0x7FFFu + ((u >> 16) & 1u);          // round-to-nearest-even
  unsigned short h = (unsigned short)(u >> 16);
  return __builtin_bit_cast(bf16_t, h);
}

// ---------------------------------------------------------------------------
// Generic GEMM: C[M,N] = act(A[M,K] @ W[K,N] + bias)
// block = 128 threads = 4 waves; block tile = 64(M) x 64(N).
// Wave w owns N strip [n0, n0+16); loops 4 M-subtiles with 4 independent
// f32 accumulators (4 WMMAs per K-step, no RAW chains).  A tile (64x32) and
// per-wave transposed B tile (16x32) staged in LDS as bf16 so every fragment
// is two contiguous 16B chunks -> ds_load_b128.  Next A K-tile prefetched
// via global_prefetch_b8 while the current one is computed.
// ---------------------------------------------------------------------------
__global__ __launch_bounds__(128)
void k_gemm_wmma(const float* __restrict__ A, int lda,
                 const float* __restrict__ W,               // [K,N] row-major
                 const float* __restrict__ bias,            // [N] or null
                 float* __restrict__ C, int ldc,
                 int M, int N, int K, int do_relu)
{
  __shared__ bf16_t As[64][40];        // 64 x 32 (+pad), 80B row pitch
  __shared__ bf16_t Bs[4][16][40];     // per-wave [n][k] (transposed), 16 x 32

  const int tid  = threadIdx.x;
  const int wave = tid >> 5;
  const int lane = tid & 31;
  const int m0 = blockIdx.x * 64;
  const int n0 = blockIdx.y * 64 + wave * 16;
  const int mrow = lane & 15;          // row (A) / col (B,C) within tile
  const int g    = lane >> 4;          // lane half

  v8f acc0 = {}, acc1 = {}, acc2 = {}, acc3 = {};

  for (int k0 = 0; k0 < K; k0 += 32) {
    // stage A tile 64x32 (all 128 threads, 16 elems each, lane-coalesced)
#pragma unroll
    for (int t = 0; t < 16; ++t) {
      int idx = t * 128 + tid;         // 0..2047
      int r = idx >> 5, kk = idx & 31;
      float x = 0.f;
      int gm = m0 + r, gk = k0 + kk;
      if (gm < M && gk < K) x = A[(long)gm * lda + gk];
      As[r][kk] = f2bf(x);
    }
    // stage this wave's B tile, TRANSPOSED: Bs[col][kk]
    if (n0 < N) {
#pragma unroll
      for (int t = 0; t < 16; ++t) {
        int idx = t * 32 + lane;       // 0..511
        int col = idx & 15, kk = idx >> 4;
        float x = 0.f;
        int gk = k0 + kk, gn = n0 + col;
        if (gk < K && gn < N) x = W[(long)gk * N + gn];
        Bs[wave][col][kk] = f2bf(x);
      }
    }
    // prefetch next A K-tile (gfx1250 global_prefetch_b8; no counters used)
    if (k0 + 32 < K) {
      int gm = m0 + (tid >> 1);                 // 64 rows, 2 threads per row
      int gk = k0 + 32 + (tid & 1) * 16;
      if (gm < M && gk < K)
        __builtin_prefetch(&A[(long)gm * lda + gk], 0, 3);
    }
    __syncthreads();
    if (n0 < N) {
      // B fragment: lane needs K=16g..16g+15 at column mrow -> contiguous
      v8bf b0 = *(const v8bf*)&Bs[wave][mrow][16 * g];
      v8bf b1 = *(const v8bf*)&Bs[wave][mrow][16 * g + 8];
      v16bf b = __builtin_shufflevector(b0, b1, 0,1,2,3,4,5,6,7,8,9,10,11,12,13,14,15);
      // A fragments: lane needs K=16g..16g+15 -> contiguous
#pragma unroll
      for (int sm = 0; sm < 4; ++sm) {
        const bf16_t* arow = &As[sm * 16 + mrow][0];
        v8bf a0 = *(const v8bf*)&arow[16 * g];
        v8bf a1 = *(const v8bf*)&arow[16 * g + 8];
        v16bf a = __builtin_shufflevector(a0, a1, 0,1,2,3,4,5,6,7,8,9,10,11,12,13,14,15);
        if (sm == 0) acc0 = __builtin_amdgcn_wmma_f32_16x16x32_bf16(false, a, false, b, (short)0, acc0, false, false);
        if (sm == 1) acc1 = __builtin_amdgcn_wmma_f32_16x16x32_bf16(false, a, false, b, (short)0, acc1, false, false);
        if (sm == 2) acc2 = __builtin_amdgcn_wmma_f32_16x16x32_bf16(false, a, false, b, (short)0, acc2, false, false);
        if (sm == 3) acc3 = __builtin_amdgcn_wmma_f32_16x16x32_bf16(false, a, false, b, (short)0, acc3, false, false);
      }
    }
    __syncthreads();
  }

  if (n0 >= N) return;
  int col = n0 + mrow;
  if (col >= N) return;
  float bv = bias ? bias[col] : 0.f;
  const bool fullM = (m0 + 64 <= M);
  // C/D layout: VGPR r -> row 8*g + r, col = lane&15
#pragma unroll
  for (int sm = 0; sm < 4; ++sm) {
    v8f acc = (sm == 0) ? acc0 : (sm == 1) ? acc1 : (sm == 2) ? acc2 : acc3;
    int rowbase = m0 + sm * 16 + 8 * g;
    float* cp = C + (long)rowbase * ldc + col;
    if (fullM) {
#pragma unroll
      for (int r = 0; r < 8; ++r) {
        float vv = acc[r] + bv;
        if (do_relu) vv = fmaxf(vv, 0.f);
        cp[(long)r * ldc] = vv;
      }
    } else {
#pragma unroll
      for (int r = 0; r < 8; ++r) {
        if (rowbase + r < M) {
          float vv = acc[r] + bv;
          if (do_relu) vv = fmaxf(vv, 0.f);
          cp[(long)r * ldc] = vv;
        }
      }
    }
  }
}

// ---------------------------------------------------------------------------
// Elementwise / data-movement kernels
// ---------------------------------------------------------------------------
__global__ void k_zero(float* p, long n) {
  long i = (long)blockIdx.x * blockDim.x + threadIdx.x;
  if (i < n) p[i] = 0.f;
}

__global__ void k_deg(const int* __restrict__ dst, float* deg, int E) {
  int e = blockIdx.x * blockDim.x + threadIdx.x;
  if (e < E) atomicAdd(&deg[dst[e]], 1.f);
}

__global__ void k_invdenom(float* d, int n) {
  int i = blockIdx.x * blockDim.x + threadIdx.x;
  if (i < n) d[i] = 1.f / fmaxf(d[i], 1.f);
}

// LayerNorm on scalars (+ optional residual delta scaled per-row) and the GVP
// vector RMS normalization.  One thread per row.
__global__ void k_ln(const float* __restrict__ sb, const float* __restrict__ sd,
                     const float* __restrict__ rscale,
                     const float* __restrict__ gam, const float* __restrict__ bet,
                     float* __restrict__ so, int S,
                     const float* __restrict__ vb, const float* __restrict__ vd,
                     float* __restrict__ vo, int VC, int M)
{
  int i = blockIdx.x * blockDim.x + threadIdx.x;
  if (i >= M) return;
  float sc = rscale ? rscale[i] : 1.f;
  const float* srow = sb + (long)i * S;
  const float* drow = sd ? sd + (long)i * S : nullptr;
  float* out = so + (long)i * S;
  float mu = 0.f;
  for (int c = 0; c < S; ++c) {
    float x = srow[c] + (drow ? drow[c] * sc : 0.f);
    out[c] = x; mu += x;
  }
  mu /= (float)S;
  float var = 0.f;
  for (int c = 0; c < S; ++c) { float d = out[c] - mu; var += d * d; }
  var /= (float)S;
  float inv = rsqrtf(var + 1e-5f);
  for (int c = 0; c < S; ++c) out[c] = (out[c] - mu) * inv * gam[c] + bet[c];

  if (VC > 0) {
    const float* vrow = vb + (long)i * 3 * VC;
    const float* vdr  = vd ? vd + (long)i * 3 * VC : nullptr;
    float* vout = vo + (long)i * 3 * VC;
    float mean = 0.f;
    for (int c = 0; c < VC; ++c) {
      float n2 = 0.f;
      for (int x = 0; x < 3; ++x) {
        float w = vrow[x * VC + c] + (vdr ? vdr[x * VC + c] * sc : 0.f);
        vout[x * VC + c] = w; n2 += w * w;
      }
      mean += fmaxf(n2, 1e-8f);
    }
    mean /= (float)VC;
    float vinv = rsqrtf(mean);
    for (int j = 0; j < 3 * VC; ++j) vout[j] *= vinv;
  }
}

// vn[i,h] = sqrt(clip(sum_xyz vh[i,xyz,h]^2)), written into dst[i*ldd+off+h]
__global__ void k_vnorm(const float* __restrict__ vh, int H,
                        float* __restrict__ dst, int ldd, int off, long M)
{
  long idx = (long)blockIdx.x * blockDim.x + threadIdx.x;
  if (idx >= M * H) return;
  long i = idx / H; int h = (int)(idx % H);
  const float* p = vh + i * 3 * H + h;
  float n2 = p[0]*p[0] + p[H]*p[H] + p[2*H]*p[2*H];
  dst[i * ldd + off + h] = sqrtf(fmaxf(n2, 1e-8f));
}

// vout *= sigmoid(||vout||_xyz) per (row, channel)
__global__ void k_vgate(float* __restrict__ v, int VC, long M)
{
  long idx = (long)blockIdx.x * blockDim.x + threadIdx.x;
  if (idx >= M * VC) return;
  long i = idx / VC; int c = (int)(idx % VC);
  float* p = v + i * 3 * VC + c;
  float n2 = p[0]*p[0] + p[VC]*p[VC] + p[2*VC]*p[2*VC];
  float s = 1.f / (1.f + expf(-sqrtf(fmaxf(n2, 1e-8f))));
  p[0] *= s; p[VC] *= s; p[2*VC] *= s;
}

__global__ void k_copy2d(const float* __restrict__ src, int lds_,
                         float* __restrict__ dst, int ldd, int off, int C, long M)
{
  long idx = (long)blockIdx.x * blockDim.x + threadIdx.x;
  if (idx >= M * C) return;
  long i = idx / C; int c = (int)(idx % C);
  dst[i * ldd + off + c] = src[i * lds_ + c];
}

// Build per-edge scalar concat [s_j | e_s | s_i] into ld-417 buffer
__global__ void k_gather_s(const float* __restrict__ s, const float* __restrict__ es,
                           const int* __restrict__ src, const int* __restrict__ dst,
                           float* __restrict__ out, int E)
{
  long idx = (long)blockIdx.x * blockDim.x + threadIdx.x;
  if (idx >= (long)E * 288) return;
  long e = idx / 288; int c = (int)(idx % 288);
  float v;
  if (c < 128)       v = s[(long)src[e] * 128 + c];
  else if (c < 160)  v = es[(long)e * 32 + (c - 128)];
  else               v = s[(long)dst[e] * 128 + (c - 160)];
  out[e * 417 + c] = v;
}

// Build per-edge vector concat [v_j | e_v | v_i] in [E,3,129] layout
__global__ void k_gather_v(const float* __restrict__ v, const float* __restrict__ ev,
                           const int* __restrict__ src, const int* __restrict__ dst,
                           float* __restrict__ out, int E)
{
  long idx = (long)blockIdx.x * blockDim.x + threadIdx.x;
  if (idx >= (long)E * 3 * 129) return;
  int c = (int)(idx % 129);
  long t = idx / 129;
  int xyz = (int)(t % 3);
  long e = t / 3;
  float x;
  if (c < 64)        x = v[(long)src[e] * 192 + xyz * 64 + c];
  else if (c == 64)  x = ev[e * 3 + xyz];
  else               x = v[(long)dst[e] * 192 + xyz * 64 + (c - 65)];
  out[e * 387 + xyz * 129 + c] = x;
}

__global__ void k_scatter_add(const float* __restrict__ m, const int* __restrict__ dst,
                              float* __restrict__ acc, int C, int E)
{
  long idx = (long)blockIdx.x * blockDim.x + threadIdx.x;
  if (idx >= (long)E * C) return;
  long e = idx / C; int c = (int)(idx % C);
  atomicAdd(&acc[(long)dst[e] * C + c], m[(long)e * C + c]);
}

// Edge-embed vector path (h = vo = 1): vh = ev0*wh, vn -> s_in col 16, ev = vh*wv
__global__ void k_edge_vec(const float* __restrict__ ev0, const float* __restrict__ wh,
                           const float* __restrict__ wv,
                           float* __restrict__ s_in, float* __restrict__ ev_out, int E)
{
  int e = blockIdx.x * blockDim.x + threadIdx.x;
  if (e >= E) return;
  float whv = wh[0], wvv = wv[0];
  float vx = ev0[(long)e*3+0]*whv, vy = ev0[(long)e*3+1]*whv, vz = ev0[(long)e*3+2]*whv;
  s_in[(long)e * 17 + 16] = sqrtf(fmaxf(vx*vx + vy*vy + vz*vz, 1e-8f));
  ev_out[(long)e*3+0] = vx*wvv; ev_out[(long)e*3+1] = vy*wvv; ev_out[(long)e*3+2] = vz*wvv;
}

__global__ void k_pool(const float* __restrict__ s, const int* __restrict__ batch,
                       float* __restrict__ out, int C, int N)
{
  long idx = (long)blockIdx.x * blockDim.x + threadIdx.x;
  if (idx >= (long)N * C) return;
  long n = idx / C; int c = (int)(idx % C);
  atomicAdd(&out[(long)batch[n] * C + c], s[n * C + c]);
}

// ---------------------------------------------------------------------------
// Host orchestration
// ---------------------------------------------------------------------------
static inline int nblk(long n, int bs) { return (int)((n + bs - 1) / bs); }

extern "C" void kernel_launch(void* const* d_in, const int* in_sizes, int n_in,
                              void* d_out, int out_size, void* d_ws, size_t ws_size,
                              hipStream_t stream)
{
  (void)in_sizes; (void)n_in; (void)out_size; (void)ws_size;
  const int N = NNODES, E = NEDGES, G = NGRAPH;

  const float* node_s = (const float*)d_in[0];    // [N,66]
  const float* node_v = (const float*)d_in[1];    // [N,1,3] == [N,3,1]
  const float* edge_s = (const float*)d_in[2];    // [E,16]
  const float* edge_v = (const float*)d_in[3];    // [E,1,3]
  const int*   eidx   = (const int*)d_in[4];      // [2,E]
  const int*   batch  = (const int*)d_in[5];      // [N]
  const int* src = eidx, * dst = eidx + E;

  // params flattened in jax tree order (sorted dict keys):
  // layers(l): +0..3 ff0{wh,ws_b,ws_w,wv} +4..7 ff1 +8,9 ln0{b,g} +10,11 ln1{b,g}
  //            +12..15 msg0 +16..19 msg1 +20..23 msg2
  const float* P[96];
  for (int i = 6; i < 95; ++i) P[i] = (const float*)d_in[i];
  const float* outg_wh = P[78]; const float* outg_b = P[79]; const float* outg_w = P[80];
  const float* outln_b = P[81]; const float* outln_g = P[82];
  const float* weg_wh = P[83]; const float* weg_b = P[84];
  const float* weg_w  = P[85]; const float* weg_wv = P[86];
  const float* weln_b = P[87]; const float* weln_g = P[88];
  const float* wvg_wh = P[89]; const float* wvg_b = P[90];
  const float* wvg_w  = P[91]; const float* wvg_wv = P[92];
  const float* wvln_b = P[93]; const float* wvln_g = P[94];

  // scratch bump allocator
  char* wp = (char*)d_ws;
  auto alloc = [&](long nf) {
    float* p = (float*)wp;
    wp += ((nf * 4 + 255) / 256) * 256;
    return p;
  };
  float* sA     = alloc((long)N * 128);
  float* sB     = alloc((long)N * 128);
  float* vA     = alloc((long)N * 192);
  float* vB     = alloc((long)N * 192);
  float* esb    = alloc((long)E * 32);
  float* evb    = alloc((long)E * 3);
  float* dinv   = alloc(N);
  float* s0     = alloc((long)N * 66);
  float* v0     = alloc((long)N * 3);
  float* sin_n  = alloc((long)N * 640);
  float* vh_n   = alloc((long)N * 384);
  float* fsbuf  = alloc((long)N * 512);
  float* fvbuf  = alloc((long)N * 384);
  float* ds_acc = alloc((long)N * 128);
  float* dv_acc = alloc((long)N * 192);
  float* sin_e  = alloc((long)E * 417);
  float* vcat   = alloc((long)E * 387);
  float* vh_e   = alloc((long)E * 387);
  float* msA    = alloc((long)E * 128);
  float* mvA    = alloc((long)E * 192);
  float* msB    = alloc((long)E * 128);
  float* mvB    = alloc((long)E * 192);
  float* es0    = alloc((long)E * 16);
  float* ev0b   = alloc((long)E * 3);

  auto gemm = [&](const float* A, int lda, const float* W, const float* bias,
                  float* C, int ldc, int M, int Nn, int K, int relu) {
    dim3 grid((M + 63) / 64, (Nn + 63) / 64);
    k_gemm_wmma<<<grid, 128, 0, stream>>>(A, lda, W, bias, C, ldc, M, Nn, K, relu);
  };

  // ---- degrees ----
  k_zero<<<nblk(N,256),256,0,stream>>>(dinv, N);
  k_deg<<<nblk(E,256),256,0,stream>>>(dst, dinv, E);
  k_invdenom<<<nblk(N,256),256,0,stream>>>(dinv, N);

  // ---- node input embed: LN + GVP(66,1 -> 128,64), h=64 ----
  k_ln<<<nblk(N,256),256,0,stream>>>(node_s, nullptr, nullptr, wvln_g, wvln_b,
                                     s0, 66, node_v, nullptr, v0, 1, N);
  gemm(v0, 1, wvg_wh, nullptr, vh_n, 64, 3*N, 64, 1, 0);          // vh [N,3,64]
  k_copy2d<<<nblk((long)N*66,256),256,0,stream>>>(s0, 66, sin_n, 130, 0, 66, N);
  k_vnorm<<<nblk((long)N*64,256),256,0,stream>>>(vh_n, 64, sin_n, 130, 66, N);
  gemm(sin_n, 130, wvg_w, wvg_b, sA, 128, N, 128, 130, 0);        // s [N,128]
  gemm(vh_n, 64, wvg_wv, nullptr, vA, 64, 3*N, 64, 64, 0);        // v [N,3,64]

  // ---- edge input embed: LN + GVP(16,1 -> 32,1), h=1 ----
  k_ln<<<nblk(E,256),256,0,stream>>>(edge_s, nullptr, nullptr, weln_g, weln_b,
                                     es0, 16, edge_v, nullptr, ev0b, 1, E);
  k_copy2d<<<nblk((long)E*16,256),256,0,stream>>>(es0, 16, sin_e, 17, 0, 16, E);
  k_edge_vec<<<nblk(E,256),256,0,stream>>>(ev0b, weg_wh, weg_wv, sin_e, evb, E);
  gemm(sin_e, 17, weg_w, weg_b, esb, 32, E, 32, 17, 0);           // es [E,32]

  float* s = sA; float* s_alt = sB;
  float* v = vA; float* v_alt = vB;

  for (int l = 0; l < 3; ++l) {
    int L = 6 + 24 * l;
    const float *ff0_wh=P[L+0], *ff0_b=P[L+1], *ff0_w=P[L+2], *ff0_wv=P[L+3];
    const float *ff1_wh=P[L+4], *ff1_b=P[L+5], *ff1_w=P[L+6], *ff1_wv=P[L+7];
    const float *ln0_b=P[L+8], *ln0_g=P[L+9], *ln1_b=P[L+10], *ln1_g=P[L+11];
    const float *m0_wh=P[L+12], *m0_b=P[L+13], *m0_w=P[L+14], *m0_wv=P[L+15];
    const float *m1_wh=P[L+16], *m1_b=P[L+17], *m1_w=P[L+18], *m1_wv=P[L+19];
    const float *m2_wh=P[L+20], *m2_b=P[L+21], *m2_w=P[L+22], *m2_wv=P[L+23];

    // ---- message GVP 0: (288,129) -> (128,64), h=129, relu/sig ----
    k_gather_s<<<nblk((long)E*288,256),256,0,stream>>>(s, esb, src, dst, sin_e, E);
    k_gather_v<<<nblk((long)E*387,256),256,0,stream>>>(v, evb, src, dst, vcat, E);
    gemm(vcat, 129, m0_wh, nullptr, vh_e, 129, 3*E, 129, 129, 0);
    k_vnorm<<<nblk((long)E*129,256),256,0,stream>>>(vh_e, 129, sin_e, 417, 288, E);
    gemm(sin_e, 417, m0_w, m0_b, msA, 128, E, 128, 417, 1);
    gemm(vh_e, 129, m0_wv, nullptr, mvA, 64, 3*E, 64, 129, 0);
    k_vgate<<<nblk((long)E*64,256),256,0,stream>>>(mvA, 64, E);
    // ---- message GVP 1: (128,64) -> (128,64), relu/sig ----
    gemm(mvA, 64, m1_wh, nullptr, vh_e, 64, 3*E, 64, 64, 0);
    k_copy2d<<<nblk((long)E*128,256),256,0,stream>>>(msA, 128, sin_e, 192, 0, 128, E);
    k_vnorm<<<nblk((long)E*64,256),256,0,stream>>>(vh_e, 64, sin_e, 192, 128, E);
    gemm(sin_e, 192, m1_w, m1_b, msB, 128, E, 128, 192, 1);
    gemm(vh_e, 64, m1_wv, nullptr, mvB, 64, 3*E, 64, 64, 0);
    k_vgate<<<nblk((long)E*64,256),256,0,stream>>>(mvB, 64, E);
    // ---- message GVP 2: no activations ----
    gemm(mvB, 64, m2_wh, nullptr, vh_e, 64, 3*E, 64, 64, 0);
    k_copy2d<<<nblk((long)E*128,256),256,0,stream>>>(msB, 128, sin_e, 192, 0, 128, E);
    k_vnorm<<<nblk((long)E*64,256),256,0,stream>>>(vh_e, 64, sin_e, 192, 128, E);
    gemm(sin_e, 192, m2_w, m2_b, msA, 128, E, 128, 192, 0);
    gemm(vh_e, 64, m2_wv, nullptr, mvA, 64, 3*E, 64, 64, 0);
    // ---- aggregate (segment_sum / deg) + residual + LN ----
    k_zero<<<nblk((long)N*128,256),256,0,stream>>>(ds_acc, (long)N*128);
    k_zero<<<nblk((long)N*192,256),256,0,stream>>>(dv_acc, (long)N*192);
    k_scatter_add<<<nblk((long)E*128,256),256,0,stream>>>(msA, dst, ds_acc, 128, E);
    k_scatter_add<<<nblk((long)E*192,256),256,0,stream>>>(mvA, dst, dv_acc, 192, E);
    k_ln<<<nblk(N,256),256,0,stream>>>(s, ds_acc, dinv, ln0_g, ln0_b, s_alt, 128,
                                       v, dv_acc, v_alt, 64, N);
    { float* t = s; s = s_alt; s_alt = t; t = v; v = v_alt; v_alt = t; }

    // ---- FF GVP 0: (128,64) -> (512,128), h=128, relu/sig ----
    gemm(v, 64, ff0_wh, nullptr, vh_n, 128, 3*N, 128, 64, 0);
    k_copy2d<<<nblk((long)N*128,256),256,0,stream>>>(s, 128, sin_n, 256, 0, 128, N);
    k_vnorm<<<nblk((long)N*128,256),256,0,stream>>>(vh_n, 128, sin_n, 256, 128, N);
    gemm(sin_n, 256, ff0_w, ff0_b, fsbuf, 512, N, 512, 256, 1);
    gemm(vh_n, 128, ff0_wv, nullptr, fvbuf, 128, 3*N, 128, 128, 0);
    k_vgate<<<nblk((long)N*128,256),256,0,stream>>>(fvbuf, 128, N);
    // ---- FF GVP 1: (512,128) -> (128,64), h=128 ----
    gemm(fvbuf, 128, ff1_wh, nullptr, vh_n, 128, 3*N, 128, 128, 0);
    k_copy2d<<<nblk((long)N*512,256),256,0,stream>>>(fsbuf, 512, sin_n, 640, 0, 512, N);
    k_vnorm<<<nblk((long)N*128,256),256,0,stream>>>(vh_n, 128, sin_n, 640, 512, N);
    gemm(sin_n, 640, ff1_w, ff1_b, ds_acc, 128, N, 128, 640, 0);   // fs2
    gemm(vh_n, 128, ff1_wv, nullptr, dv_acc, 64, 3*N, 64, 128, 0); // fv2
    k_ln<<<nblk(N,256),256,0,stream>>>(s, ds_acc, nullptr, ln1_g, ln1_b, s_alt, 128,
                                       v, dv_acc, v_alt, 64, N);
    { float* t = s; s = s_alt; s_alt = t; t = v; v = v_alt; v_alt = t; }
  }

  // ---- output: LN + GVP(128,64 -> 128, vo=0), relu on s ----
  k_ln<<<nblk(N,256),256,0,stream>>>(s, nullptr, nullptr, outln_g, outln_b, s_alt, 128,
                                     v, nullptr, v_alt, 64, N);
  gemm(v_alt, 64, outg_wh, nullptr, vh_n, 64, 3*N, 64, 64, 0);
  k_copy2d<<<nblk((long)N*128,256),256,0,stream>>>(s_alt, 128, sin_n, 192, 0, 128, N);
  k_vnorm<<<nblk((long)N*64,256),256,0,stream>>>(vh_n, 64, sin_n, 192, 128, N);
  gemm(sin_n, 192, outg_w, outg_b, fsbuf, 128, N, 128, 192, 1);

  // ---- global add pool ----
  k_zero<<<nblk((long)G*128,256),256,0,stream>>>((float*)d_out, (long)G*128);
  k_pool<<<nblk((long)N*128,256),256,0,stream>>>(fsbuf, batch, (float*)d_out, 128, N);
}
